// SimpleLlamaAttention_85727547228329
// MI455X (gfx1250) — matrix operand order repaired and checked by reading, not compiled, not run on hardware
//
#include <hip/hip_runtime.h>
#include <hip/hip_bf16.h>
#include <math.h>

// ---------------------------------------------------------------------------
// CDNA5 (gfx1250) wave32 WMMA types and helpers
// ---------------------------------------------------------------------------
typedef __attribute__((ext_vector_type(16))) __bf16 v16bf;
typedef __attribute__((ext_vector_type(8)))  __bf16 v8bf;
typedef __attribute__((ext_vector_type(4)))  __bf16 v4bf;
typedef __attribute__((ext_vector_type(8)))  float  v8f;

__device__ __forceinline__ v8f wmma_bf16(v16bf a, v16bf b, v8f c) {
    // (neg_a, A, neg_b, B, c_mod, C, reuse_a, reuse_b)
    return __builtin_amdgcn_wmma_f32_16x16x32_bf16(false, a, false, b,
                                                   (short)0, c, false, false);
}

__device__ __forceinline__ v16bf cat8(v8bf lo, v8bf hi) {
    return __builtin_shufflevector(lo, hi, 0, 1, 2, 3, 4, 5, 6, 7,
                                   8, 9, 10, 11, 12, 13, 14, 15);
}
__device__ __forceinline__ v4bf cvt4(float4 a) {
    v4bf r;
    r[0] = (__bf16)a.x; r[1] = (__bf16)a.y; r[2] = (__bf16)a.z; r[3] = (__bf16)a.w;
    return r;
}
__device__ __forceinline__ v8bf cvt8(float4 a, float4 b) {
    v8bf r;
    r[0] = (__bf16)a.x; r[1] = (__bf16)a.y; r[2] = (__bf16)a.z; r[3] = (__bf16)a.w;
    r[4] = (__bf16)b.x; r[5] = (__bf16)b.y; r[6] = (__bf16)b.z; r[7] = (__bf16)b.w;
    return r;
}
__device__ __forceinline__ v16bf cvt16(float4 a, float4 b, float4 c, float4 d) {
    return cat8(cvt8(a, b), cvt8(c, d));
}

// A-fragment (16-bit 16x32): lane element i -> k = half*16 + i (one 16-chunk).
// B-fragment (16-bit 32x16): lane element i -> k = half*16 + i (one 16-chunk).

// Build B-fragment from 16 consecutive fp32 at row[half*16 .. half*16+15].
__device__ __forceinline__ v16bf bfrag_from_f32row(const float* row, int half) {
    const float4* p = (const float4*)row;
    int b0 = half * 4;
    return cvt16(p[b0], p[b0 + 1], p[b0 + 2], p[b0 + 3]);
}
// Build A-fragment from fp32 row: 16 consecutive at [half*16].
__device__ __forceinline__ v16bf afrag_from_f32row(const float* row, int half) {
    const float4* p = (const float4*)row;
    int b0 = half * 4;
    return cvt16(p[b0], p[b0 + 1], p[b0 + 2], p[b0 + 3]);
}

// ---------------------------------------------------------------------------
// Kernel 1: C = A[M,K] @ W[N,K]^T  (fp32 in/out, bf16 WMMA, f32 accumulate)
// 256 threads = 8 waves; 128x128 block tile, K stepped by 32.
// Wave grid 4(M) x 2(N): each wave computes 32x64 via 8 WMMAs per K-step.
// transC = 1 stores C transposed per batch: C[b][n][s]  (row = b*S + s).
// ---------------------------------------------------------------------------
#define GT_M 128
#define GT_N 128
#define GT_K 32

__global__ __launch_bounds__(256) void gemm_xwt_bf16(
    const float* __restrict__ A, const float* __restrict__ W,
    float* __restrict__ C, int M, int N, int K, int transC, int S)
{
    __shared__ __bf16 As[GT_M][GT_K];   // 8 KB
    __shared__ __bf16 Ws[GT_N][GT_K];   // 8 KB

    const int bm   = blockIdx.y * GT_M;
    const int bn   = blockIdx.x * GT_N;
    const int tid  = threadIdx.x;
    const int lane = tid & 31;
    const int wid  = tid >> 5;
    const int wrow = (wid & 3) * 32;    // 0,32,64,96
    const int wcol = (wid >> 2) * 64;   // 0,64
    const int m    = lane & 15;
    const int half = lane >> 4;

    v8f acc[2][4];
    #pragma unroll
    for (int ai = 0; ai < 2; ++ai)
        #pragma unroll
        for (int bi = 0; bi < 4; ++bi)
            acc[ai][bi] = (v8f){};

    for (int k0 = 0; k0 < K; k0 += GT_K) {
        // ---- staging: 128x32 fp32 -> bf16 per matrix; 4 float4 per thread ----
        float4 fa[4], fw[4];
        #pragma unroll
        for (int it = 0; it < 4; ++it) {
            int idx = tid + it * 256;
            int r = idx >> 3, c = idx & 7;      // row, float4-chunk
            fa[it] = *(const float4*)(A + (size_t)(bm + r) * K + (k0 + c * 4));
            fw[it] = *(const float4*)(W + (size_t)(bn + r) * K + (k0 + c * 4));
        }
        #pragma unroll
        for (int it = 0; it < 4; ++it) {
            int idx = tid + it * 256;
            int r = idx >> 3, c = idx & 7;
            *(v4bf*)&As[r][c * 4] = cvt4(fa[it]);
            *(v4bf*)&Ws[r][c * 4] = cvt4(fw[it]);
        }
        if (k0 + GT_K < K) {                     // global_prefetch next tile
            int r = tid >> 3, c = tid & 7;
            __builtin_prefetch(A + (size_t)(bm + r) * K + (k0 + GT_K + c * 4));
            __builtin_prefetch(W + (size_t)(bn + r) * K + (k0 + GT_K + c * 4));
        }
        __syncthreads();

        // ---- fragments from LDS (all contiguous b128 loads) ----
        v16bf af[2], bfv[4];
        #pragma unroll
        for (int ai = 0; ai < 2; ++ai) {
            const __bf16* row = &As[wrow + ai * 16 + m][0];
            v8bf lo = *(const v8bf*)(row + half * 16);
            v8bf hi = *(const v8bf*)(row + half * 16 + 8);
            af[ai] = cat8(lo, hi);
        }
        #pragma unroll
        for (int bi = 0; bi < 4; ++bi)
            bfv[bi] = *(const v16bf*)&Ws[wcol + bi * 16 + m][half * 16];

        #pragma unroll
        for (int ai = 0; ai < 2; ++ai)
            #pragma unroll
            for (int bi = 0; bi < 4; ++bi)
                acc[ai][bi] = wmma_bf16(af[ai], bfv[bi], acc[ai][bi]);
        __syncthreads();
    }

    // ---- epilogue: C/D layout lane->col = m, VGPR r -> row = r + 8*half ----
    #pragma unroll
    for (int ai = 0; ai < 2; ++ai) {
        #pragma unroll
        for (int bi = 0; bi < 4; ++bi) {
            #pragma unroll
            for (int r = 0; r < 8; ++r) {
                int row = bm + wrow + ai * 16 + r + 8 * half;
                int col = bn + wcol + bi * 16 + m;
                if (!transC) {
                    C[(size_t)row * N + col] = acc[ai][bi][r];
                } else {
                    int bb = row / S, ss = row % S;
                    C[((size_t)bb * N + col) * S + ss] = acc[ai][bi][r];
                }
            }
        }
    }
}

// ---------------------------------------------------------------------------
// Kernel 2: RoPE in-place on [tokens, heads*64]; one thread per (d, d+32) pair.
// ---------------------------------------------------------------------------
__global__ void rope_kernel(float* __restrict__ X, const int* __restrict__ pos_ids,
                            int tokens, int heads)
{
    int p = blockIdx.x * blockDim.x + threadIdx.x;
    int total = tokens * heads * 32;
    if (p >= total) return;
    int d  = p & 31;
    int th = p >> 5;          // token*heads + head
    int t  = th / heads;
    float posf = (float)pos_ids[t];
    float inv_freq = __powf(10000.0f, -((float)(2 * d) / 64.0f));
    float ang = posf * inv_freq;
    float s, c;
    __sincosf(ang, &s, &c);
    float* base = X + (size_t)th * 64;
    float x1 = base[d];
    float x2 = base[d + 32];
    base[d]      = x1 * c - x2 * s;
    base[d + 32] = x2 * c + x1 * s;
}

// ---------------------------------------------------------------------------
// Kernel 3: causal GQA flash attention; one wave per 16-query tile.
// grid = (S/16, B*32). Keys streamed in chunks of 32 (full WMMA K dim).
// Q,K: [token][heads*64] fp32 (post-RoPE).  Vt: [B][512][S] fp32 (transposed).
// ---------------------------------------------------------------------------
__global__ __launch_bounds__(32) void flash_attn_kernel(
    const float* __restrict__ Q, const float* __restrict__ K,
    const float* __restrict__ Vt, float* __restrict__ O, int S)
{
    const int qt   = blockIdx.x;
    const int bh   = blockIdx.y;
    const int b    = bh >> 5;
    const int h    = bh & 31;
    const int kvh  = h >> 2;            // GQA 4:1
    const int lane = threadIdx.x;
    const int m    = lane & 15;
    const int half = lane >> 4;

    const size_t qtok0 = (size_t)b * S + (size_t)qt * 16;
    const float* q  = Q + qtok0 * 2048 + h * 64;
    const float* kp = K + (size_t)b * S * 512 + kvh * 64;
    const float* vt = Vt + (size_t)b * 512 * S + (size_t)kvh * 64 * S;

    // Q fragments (A 16x32) for d=0..31 and d=32..63 — vectorized f32 loads.
    const float* qrow = q + (size_t)m * 2048;
    v16bf qa0 = afrag_from_f32row(qrow, half);
    v16bf qa1 = afrag_from_f32row(qrow + 32, half);

    float mrow[8], lrow[8];
    v8f o0 = {}, o1 = {}, o2 = {}, o3 = {};
    #pragma unroll
    for (int r = 0; r < 8; ++r) { mrow[r] = -1e30f; lrow[r] = 0.0f; }

    __shared__ __bf16 Ps[16][32];       // probs, [q-row][key-in-chunk]

    const float scale = 0.125f;         // 1/sqrt(64)
    const int   kv_end = qt * 16 + 16;  // causal bound

    for (int kb = 0; kb < kv_end; kb += 32) {
        // ---- scores: contiguous 16-float B-frag loads from K rows ----
        const float* krow0 = kp + (size_t)(kb + m) * 512;
        const float* krow1 = kp + (size_t)(kb + 16 + m) * 512;
        v16bf kb0l = bfrag_from_f32row(krow0, half);
        v16bf kb0h = bfrag_from_f32row(krow0 + 32, half);
        v16bf kb1l = bfrag_from_f32row(krow1, half);
        v16bf kb1h = bfrag_from_f32row(krow1 + 32, half);

        v8f s0 = {}, s1 = {};
        s0 = wmma_bf16(qa0, kb0l, s0);
        s0 = wmma_bf16(qa1, kb0h, s0);
        s1 = wmma_bf16(qa0, kb1l, s1);
        s1 = wmma_bf16(qa1, kb1h, s1);

        // ---- causal mask + online softmax ----
        float alpha8[8];
        #pragma unroll
        for (int r = 0; r < 8; ++r) {
            int row_g = qt * 16 + r + 8 * half;
            float v0 = s0[r] * scale;
            float v1 = s1[r] * scale;
            if (kb + m > row_g)      v0 = -1e30f;
            if (kb + 16 + m > row_g) v1 = -1e30f;

            float mx = fmaxf(v0, v1);
            #pragma unroll
            for (int off = 1; off < 16; off <<= 1)
                mx = fmaxf(mx, __shfl_xor(mx, off, 32));
            float mnew  = fmaxf(mrow[r], mx);
            float alpha = __expf(mrow[r] - mnew);
            float p0 = __expf(v0 - mnew);
            float p1 = __expf(v1 - mnew);
            float ssum = p0 + p1;
            #pragma unroll
            for (int off = 1; off < 16; off <<= 1)
                ssum += __shfl_xor(ssum, off, 32);
            lrow[r] = lrow[r] * alpha + ssum;
            mrow[r] = mnew;
            alpha8[r] = alpha;
            Ps[r + 8 * half][m]      = (__bf16)p0;
            Ps[r + 8 * half][16 + m] = (__bf16)p1;
        }
        #pragma unroll
        for (int r = 0; r < 8; ++r) {
            o0[r] *= alpha8[r]; o1[r] *= alpha8[r];
            o2[r] *= alpha8[r]; o3[r] *= alpha8[r];
        }
        __syncthreads();   // cross-lane P transpose through LDS

        // ---- P as A-fragment (two contiguous b128 chunks per lane) ----
        const __bf16* prow = &Ps[m][0];
        v16bf pa = cat8(*(const v8bf*)(prow + half * 16),
                        *(const v8bf*)(prow + half * 16 + 8));

        // ---- P.V with V transposed: contiguous 16-float loads per dim-row ----
        v16bf vb0 = bfrag_from_f32row(vt + (size_t)(0  + m) * S + kb, half);
        v16bf vb1 = bfrag_from_f32row(vt + (size_t)(16 + m) * S + kb, half);
        v16bf vb2 = bfrag_from_f32row(vt + (size_t)(32 + m) * S + kb, half);
        v16bf vb3 = bfrag_from_f32row(vt + (size_t)(48 + m) * S + kb, half);

        o0 = wmma_bf16(pa, vb0, o0);
        o1 = wmma_bf16(pa, vb1, o1);
        o2 = wmma_bf16(pa, vb2, o2);
        o3 = wmma_bf16(pa, vb3, o3);
        __syncthreads();
    }

    // ---- finalize: divide by row sums, write [token, h*64 + d] ----
    #pragma unroll
    for (int r = 0; r < 8; ++r) {
        float inv = 1.0f / lrow[r];
        size_t row = qtok0 + (size_t)(r + 8 * half);
        float* op = O + row * 2048 + h * 64 + m;
        op[0]  = o0[r] * inv;
        op[16] = o1[r] * inv;
        op[32] = o2[r] * inv;
        op[48] = o3[r] * inv;
    }
}

// ---------------------------------------------------------------------------
// Host-side launch
// ---------------------------------------------------------------------------
extern "C" void kernel_launch(void* const* d_in, const int* in_sizes, int n_in,
                              void* d_out, int out_size, void* d_ws, size_t ws_size,
                              hipStream_t stream) {
    const float* X   = (const float*)d_in[0];   // [B,S,2048]
    const int*   pos = (const int*)d_in[1];     // [B,S]
    const float* Wq  = (const float*)d_in[2];   // [2048,2048]
    const float* Wk  = (const float*)d_in[3];   // [512,2048]
    const float* Wv  = (const float*)d_in[4];   // [512,2048]
    const float* Wo  = (const float*)d_in[5];   // [2048,2048]
    float* out = (float*)d_out;

    const int HID = 2048, S = 2048, NH = 32, NKV = 8, D = 64;
    const int tokens = in_sizes[0] / HID;       // B*S = 4096
    const int B = tokens / S;
    const int KVW = NKV * D;                    // 512

    char* ws = (char*)d_ws;
    float* Qb  = (float*)ws;                                          // 32 MB
    float* Kb  = (float*)(ws + (size_t)tokens * HID * 4);             //  8 MB
    float* Vtb = (float*)(ws + (size_t)tokens * HID * 4
                             + (size_t)tokens * KVW * 4);             //  8 MB (transposed)
    float* Ab  = (float*)(ws + (size_t)tokens * HID * 4
                             + 2 * (size_t)tokens * KVW * 4);         // 32 MB

    dim3 blk(256);
    // QKV projections (V stored transposed per batch: [B][512][S])
    gemm_xwt_bf16<<<dim3(HID / GT_N, tokens / GT_M), blk, 0, stream>>>(X, Wq, Qb, tokens, HID, HID, 0, S);
    gemm_xwt_bf16<<<dim3(KVW / GT_N, tokens / GT_M), blk, 0, stream>>>(X, Wk, Kb, tokens, KVW, HID, 0, S);
    gemm_xwt_bf16<<<dim3(KVW / GT_N, tokens / GT_M), blk, 0, stream>>>(X, Wv, Vtb, tokens, KVW, HID, 1, S);

    // RoPE on Q and K
    int nq = tokens * NH * 32;
    rope_kernel<<<(nq + 255) / 256, 256, 0, stream>>>(Qb, pos, tokens, NH);
    int nk = tokens * NKV * 32;
    rope_kernel<<<(nk + 255) / 256, 256, 0, stream>>>(Kb, pos, tokens, NKV);

    // Causal GQA flash attention
    flash_attn_kernel<<<dim3(S / 16, B * NH), dim3(32), 0, stream>>>(Qb, Kb, Vtb, Ab, S);

    // Output projection
    gemm_xwt_bf16<<<dim3(HID / GT_N, tokens / GT_M), blk, 0, stream>>>(Ab, Wo, out, tokens, HID, HID, 0, S);
}
